// GineEncoder_67113158967916
// MI455X (gfx1250) — matrix operand order repaired and checked by reading, not compiled, not run on hardware
//
#include <hip/hip_runtime.h>

// ---------------------------------------------------------------------------
// GINE encoder for MI455X (gfx1250, wave32, WMMA).
// All GEMM operands pre-packed to bf16 with padded shapes so the WMMA inner
// loop is guard-free b128 traffic: global_load_b128 -> ds_store_b128 ->
// ds_load_b128 -> v_wmma_f32_16x16x32_bf16, double-buffered LDS, K unrolled.
// ---------------------------------------------------------------------------

typedef __attribute__((ext_vector_type(16))) __bf16 v16bf;
typedef __attribute__((ext_vector_type(8)))  __bf16 v8bf;
typedef __attribute__((ext_vector_type(8)))  float  v8f;

static constexpr int FD     = 300;  // feature dim
static constexpr int FDP    = 320;  // padded K / padded cols (10 x 32)
static constexpr int KCH    = 10;   // K chunks of 32
static constexpr int LAYERS = 5;

static inline int pad128(int m) { return (m + 127) & ~127; }

// ---------------------------------------------------------------------------
// WMMA GEMM: Out = act( Ap[M,320]bf16 @ Wp[320,320]bf16(n-major) + bias
//                       [+ addTab[gidx[row]]] )
// Block 256 thr = 8 waves; tile 128 rows x 64 cols; per wave 16 rows x 4 tiles.
// OutF: optional fp32 [M,300].  OutP: optional packed bf16 [Mpad,320].
// ---------------------------------------------------------------------------
__global__ __launch_bounds__(256)
void wmma_gemm_kernel(const __bf16* __restrict__ Ap, const __bf16* __restrict__ Wp,
                      const float* __restrict__ bias,
                      float* __restrict__ OutF, __bf16* __restrict__ OutP,
                      int M, int do_relu,
                      const float* __restrict__ addTab, const int* __restrict__ gidx)
{
    __shared__ __attribute__((aligned(16))) __bf16 lA[2][128][32]; // [row][k]
    __shared__ __attribute__((aligned(16))) __bf16 lB[2][64][32];  // [n][k]

    const int t    = threadIdx.x;
    const int lane = t & 31;
    const int wave = t >> 5;       // 0..7
    const int lm   = lane & 15;
    const int hl   = lane >> 4;

    const int R0 = blockIdx.x * 128;   // grid.x = Mpad/128 (rows zero-padded)
    const int C0 = blockIdx.y * 64;    // grid.y = 5 -> cols 0..319 (padded)

    v8f acc[4] = {};

    // unguarded staging: Ap rows/cols and Wp are fully padded
    auto stageA = [&](int kc, int bi) {
        const int r = t >> 1, seg = (t & 1) * 16;
        const __bf16* s = Ap + (long)(R0 + r) * FDP + kc * 32 + seg;
        *(v8bf*)&lA[bi][r][seg]     = *(const v8bf*)(s);
        *(v8bf*)&lA[bi][r][seg + 8] = *(const v8bf*)(s + 8);
    };
    auto stageB = [&](int kc, int bi) {
        const int n = t >> 2, seg = (t & 3) * 8;
        const __bf16* s = Wp + (long)(C0 + n) * FDP + kc * 32 + seg;
        *(v8bf*)&lB[bi][n][seg] = *(const v8bf*)(s);
    };

    stageA(0, 0);
    stageB(0, 0);
    __syncthreads();

    #pragma unroll
    for (int kc = 0; kc < KCH; ++kc) {
        const int cur = kc & 1;
        if (kc + 1 < KCH) {          // prefetch next chunk into other buffer
            stageA(kc + 1, cur ^ 1);
            stageB(kc + 1, cur ^ 1);
        }

        // A fragment (ISA 16-bit A 16x32): elems 0..7 -> K=hl*8+i, 8..15 -> 16+hl*8+i
        const __bf16* pa = &lA[cur][wave * 16 + lm][hl * 8];
        v8bf a0 = *(const v8bf*)(pa);
        v8bf a1 = *(const v8bf*)(pa + 16);
        v16bf af = __builtin_shufflevector(a0, a1,
                       0,1,2,3,4,5,6,7,8,9,10,11,12,13,14,15);

        #pragma unroll
        for (int nt = 0; nt < 4; ++nt) {
            const __bf16* pb = &lB[cur][nt * 16 + lm][hl * 8];
            v8bf b0 = *(const v8bf*)(pb);
            v8bf b1 = *(const v8bf*)(pb + 16);
            v16bf bfv = __builtin_shufflevector(b0, b1,
                            0,1,2,3,4,5,6,7,8,9,10,11,12,13,14,15);
            acc[nt] = __builtin_amdgcn_wmma_f32_16x16x32_bf16(
                          false, af, false, bfv, (short)0, acc[nt], false, false);
        }
        __syncthreads();
    }

    // epilogue: C/D layout -> lane lm = col, VGPR v -> row v + 8*hl
    #pragma unroll
    for (int nt = 0; nt < 4; ++nt) {
        const int col = C0 + nt * 16 + lm;
        #pragma unroll
        for (int v = 0; v < 8; ++v) {
            const int row = R0 + wave * 16 + hl * 8 + v;
            const bool live = (row < M) && (col < FD);
            float val = 0.0f;
            if (live) {
                val = acc[nt][v] + bias[col];
                if (gidx) val += addTab[(long)gidx[row] * FD + col];
                if (do_relu) val = fmaxf(val, 0.0f);
            }
            if (OutF && live) OutF[(long)row * FD + col] = val;
            if (OutP) OutP[(long)row * FDP + col] = (__bf16)val; // pads -> 0
        }
    }
}

// ---------------------------------------------------------------------------
// Packing / elementwise kernels
// ---------------------------------------------------------------------------

// W[300,300] (k-major fp32) -> Wp[320,320] (n-major bf16, zero-padded)
__global__ void pack_weight_kernel(const float* __restrict__ W, __bf16* __restrict__ Wp)
{
    int i = blockIdx.x * blockDim.x + threadIdx.x;
    if (i >= FDP * FDP) return;
    int n = i / FDP, k = i % FDP;
    float v = (n < FD && k < FD) ? W[k * FD + n] : 0.0f;
    Wp[i] = (__bf16)v;
}

// X[M,300] fp32 -> Xp[Mpad,320] bf16 (zero-padded), 8 cols (16B store) per thread
__global__ void pack_act_kernel(const float* __restrict__ X, __bf16* __restrict__ Xp,
                                int M, int Mpad)
{
    long i = (long)blockIdx.x * blockDim.x + threadIdx.x;
    if (i >= (long)Mpad * (FDP / 8)) return;
    int r  = (int)(i / (FDP / 8));
    int c0 = (int)(i % (FDP / 8)) * 8;
    __bf16 tmp[8];
    #pragma unroll
    for (int j = 0; j < 8; ++j) {
        int c = c0 + j;
        float v = (r < M && c < FD) ? X[(long)r * FD + c] : 0.0f;
        tmp[j] = (__bf16)v;
    }
    *(v8bf*)(Xp + (long)r * FDP + c0) = *(const v8bf*)tmp;
}

// relu([chir,fc] @ nap_w1 + b1) -> packed bf16 [Npad,320]
__global__ void node_hidden_kernel(const float* __restrict__ chir,
                                   const float* __restrict__ fc,
                                   const float* __restrict__ w1,
                                   const float* __restrict__ b1,
                                   __bf16* __restrict__ outp, int n, int npad)
{
    long i = (long)blockIdx.x * blockDim.x + threadIdx.x;
    if (i >= (long)npad * FDP) return;
    int r = (int)(i / FDP), c = (int)(i % FDP);
    float v = 0.0f;
    if (r < n && c < FD)
        v = fmaxf(chir[r] * w1[c] + fc[r] * w1[FD + c] + b1[c], 0.0f);
    outp[i] = (__bf16)v;
}

// relu(ea @ ee_w1 + b1) -> packed bf16 [Epad,320]
__global__ void edge_hidden_kernel(const float* __restrict__ ea,
                                   const float* __restrict__ w1,
                                   const float* __restrict__ b1,
                                   __bf16* __restrict__ outp, int nE, int epad)
{
    long i = (long)blockIdx.x * blockDim.x + threadIdx.x;
    if (i >= (long)epad * FDP) return;
    int e = (int)(i / FDP), c = (int)(i % FDP);
    float v = 0.0f;
    if (e < nE && c < FD)
        v = fmaxf(ea[e * 3 + 0] * w1[c] + ea[e * 3 + 1] * w1[FD + c]
                + ea[e * 3 + 2] * w1[2 * FD + c] + b1[c], 0.0f);
    outp[i] = (__bf16)v;
}

// msg = relu(h[src] + e); agg[dst] += msg  (agg pre-initialized to h -> hin)
__global__ void scatter_kernel(const float* __restrict__ h,
                               const __bf16* __restrict__ eembp,
                               const int* __restrict__ src,
                               const int* __restrict__ dst,
                               float* __restrict__ agg, int nE)
{
    int e = blockIdx.x;
    if (e >= nE) return;
    const int s = src[e], d = dst[e];
    const float*  hs = h + (long)s * FD;
    const __bf16* ee = eembp + (long)e * FDP;
    float* ag = agg + (long)d * FD;
    for (int c = threadIdx.x; c < FD; c += blockDim.x)
        atomicAdd(&ag[c], fmaxf(hs[c] + (float)ee[c], 0.0f));
}

__global__ void zero_kernel(float* __restrict__ p, long n)
{
    long i = (long)blockIdx.x * blockDim.x + threadIdx.x;
    if (i < n) p[i] = 0.0f;
}

__global__ void bn_reduce_kernel(const float* __restrict__ X,
                                 float* __restrict__ stats, int n)
{
    int c = threadIdx.x;
    if (c >= FD) return;
    float s = 0.0f, q = 0.0f;
    for (int r = blockIdx.x; r < n; r += gridDim.x) {
        float v = X[(long)r * FD + c];
        s += v; q += v * v;
    }
    atomicAdd(&stats[c], s);
    atomicAdd(&stats[FD + c], q);
}

__global__ void bn_finalize_kernel(const float* __restrict__ stats,
                                   const float* __restrict__ gamma,
                                   const float* __restrict__ beta,
                                   float* __restrict__ ss, int n)
{
    int c = threadIdx.x;
    if (c >= FD) return;
    float inv  = 1.0f / (float)n;
    float mean = stats[c] * inv;
    float var  = stats[FD + c] * inv - mean * mean;
    float sc   = gamma[c] * rsqrtf(var + 1e-5f);
    ss[c]      = sc;
    ss[FD + c] = beta[c] - mean * sc;
}

__global__ void bn_apply_kernel(const float* __restrict__ X,
                                const float* __restrict__ ss,
                                float* __restrict__ h, long total)
{
    long i = (long)blockIdx.x * blockDim.x + threadIdx.x;
    if (i >= total) return;
    int c = (int)(i % FD);
    h[i] = fmaxf(X[i] * ss[c] + ss[FD + c], 0.0f);
}

__global__ void pool_scatter_kernel(const float* __restrict__ h,
                                    const int* __restrict__ batch,
                                    float* __restrict__ sums,
                                    float* __restrict__ counts, int n)
{
    int r = blockIdx.x;
    if (r >= n) return;
    int g = batch[r];
    int c = threadIdx.x;
    if (c < FD) atomicAdd(&sums[(long)g * FD + c], h[(long)r * FD + c]);
    if (c == 0) atomicAdd(&counts[g], 1.0f);
}

// pooled mean -> packed bf16 [G,320] (G = 4096, already /128)
__global__ void pool_mean_kernel(const float* __restrict__ sums,
                                 const float* __restrict__ counts,
                                 __bf16* __restrict__ pooledp, int G)
{
    long i = (long)blockIdx.x * blockDim.x + threadIdx.x;
    if (i >= (long)G * FDP) return;
    int g = (int)(i / FDP), c = (int)(i % FDP);
    float v = 0.0f;
    if (c < FD) {
        float cnt = counts[g];
        v = (cnt > 0.0f) ? sums[(long)g * FD + c] / cnt : 0.0f;
    }
    pooledp[i] = (__bf16)v;
}

// ---------------------------------------------------------------------------
// Launch
// ---------------------------------------------------------------------------
static inline void launch_gemm(const __bf16* Ap, const __bf16* Wp, const float* bias,
                               float* OutF, __bf16* OutP, int M, int Mpad, int relu,
                               const float* addTab, const int* gidx, hipStream_t stream)
{
    dim3 grid(Mpad / 128, FDP / 64);
    wmma_gemm_kernel<<<grid, 256, 0, stream>>>(Ap, Wp, bias, OutF, OutP, M, relu,
                                               addTab, gidx);
}

extern "C" void kernel_launch(void* const* d_in, const int* in_sizes, int n_in,
                              void* d_out, int out_size, void* d_ws, size_t ws_size,
                              hipStream_t stream)
{
    (void)n_in; (void)ws_size;

    const int N = in_sizes[0];
    const int E = in_sizes[3] / 2;
    const int G = out_size / FD;
    const int Npad = pad128(N), Epad = pad128(E), Gpad = pad128(G);

    const int*   z        = (const int*)  d_in[0];
    const float* chir     = (const float*)d_in[1];
    const float* fcg      = (const float*)d_in[2];
    const int*   eidx     = (const int*)  d_in[3];
    const float* eattr    = (const float*)d_in[4];
    const int*   batch    = (const int*)  d_in[5];
    const float* atom_emb = (const float*)d_in[6];
    const float* nap_w1   = (const float*)d_in[7];
    const float* nap_b1   = (const float*)d_in[8];
    const float* nap_w2   = (const float*)d_in[9];
    const float* nap_b2   = (const float*)d_in[10];
    const float* ee_w1    = (const float*)d_in[11];
    const float* ee_b1    = (const float*)d_in[12];
    const float* ee_w2    = (const float*)d_in[13];
    const float* ee_b2    = (const float*)d_in[14];
    const float* mlp_w1   = (const float*)d_in[15];
    const float* mlp_b1   = (const float*)d_in[16];
    const float* mlp_w2   = (const float*)d_in[17];
    const float* mlp_b2   = (const float*)d_in[18];
    const float* bn_gamma = (const float*)d_in[19];
    const float* bn_beta  = (const float*)d_in[20];
    const float* pool_w   = (const float*)d_in[21];
    const float* pool_b   = (const float*)d_in[22];

    const int* src = eidx;
    const int* dst = eidx + E;

    // ---- workspace layout ----
    const size_t nd = (size_t)N * FD;
    float* h     = (float*)d_ws;                   // [N,300]
    float* buf   = h + nd;                         // hin / out2 [N,300]
    float* stats = buf + nd;                       // 600
    float* ss    = stats + 2 * FD;                 // 600
    float* sums  = ss + 2 * FD;                    // [G,300]
    float* cnts  = sums + (size_t)G * FD;          // [G]
    __bf16* wp   = (__bf16*)(cnts + G);            // 13 x [320,320]
    __bf16* eembP = wp + (size_t)13 * FDP * FDP;   // [Epad,320]
    __bf16* actA  = eembP + (size_t)Epad * FDP;    // [Epad,320] (inputs)
    __bf16* actB  = actA + (size_t)Epad * FDP;     // [Npad,320] (mid act)
    float* out   = (float*)d_out;

    __bf16* wpEE   = wp;
    __bf16* wpNAP  = wp + (size_t)1 * FDP * FDP;
    __bf16* wpM1   = wp + (size_t)2 * FDP * FDP;   // 5 matrices
    __bf16* wpM2   = wp + (size_t)7 * FDP * FDP;   // 5 matrices
    __bf16* wpPool = wp + (size_t)12 * FDP * FDP;

    const int BLK = 256;
    const int wgrid = (FDP * FDP + BLK - 1) / BLK;

    // ---- pack all 13 weight matrices (transposed, padded, bf16) ----
    pack_weight_kernel<<<wgrid, BLK, 0, stream>>>(ee_w2,  wpEE);
    pack_weight_kernel<<<wgrid, BLK, 0, stream>>>(nap_w2, wpNAP);
    for (int l = 0; l < LAYERS; ++l) {
        pack_weight_kernel<<<wgrid, BLK, 0, stream>>>(
            mlp_w1 + (size_t)l * FD * FD, wpM1 + (size_t)l * FDP * FDP);
        pack_weight_kernel<<<wgrid, BLK, 0, stream>>>(
            mlp_w2 + (size_t)l * FD * FD, wpM2 + (size_t)l * FDP * FDP);
    }
    pack_weight_kernel<<<wgrid, BLK, 0, stream>>>(pool_w, wpPool);

    // ---- edge embedding: relu(ea@w1+b1)@w2+b2 -> packed bf16 eembP ----
    {
        long tot = (long)Epad * FDP;
        edge_hidden_kernel<<<(int)((tot + BLK - 1) / BLK), BLK, 0, stream>>>(
            eattr, ee_w1, ee_b1, actA, E, Epad);
        launch_gemm(actA, wpEE, ee_b2, nullptr, eembP, E, Epad, 0,
                    nullptr, nullptr, stream);
    }

    // ---- node embedding: atom_emb[z] + relu(na@w1+b1)@w2+b2 -> h fp32 ----
    {
        long tot = (long)Npad * FDP;
        node_hidden_kernel<<<(int)((tot + BLK - 1) / BLK), BLK, 0, stream>>>(
            chir, fcg, nap_w1, nap_b1, actA, N, Npad);
        launch_gemm(actA, wpNAP, nap_b2, h, nullptr, N, Npad, 0,
                    atom_emb, z, stream);
    }

    const long ndTot = (long)nd;
    const long packN = (long)Npad * (FDP / 8);

    // ---- GINE layers ----
    for (int l = 0; l < LAYERS; ++l) {
        // hin = h + sum relu(h[src]+e)
        hipMemcpyAsync(buf, h, nd * sizeof(float), hipMemcpyDeviceToDevice, stream);
        scatter_kernel<<<E, BLK, 0, stream>>>(h, eembP, src, dst, buf, E);
        pack_act_kernel<<<(int)((packN + BLK - 1) / BLK), BLK, 0, stream>>>(
            buf, actA, N, Npad);

        // MLP: GEMM1 (relu, bf16 packed out) -> GEMM2 (fp32 out)
        launch_gemm(actA, wpM1 + (size_t)l * FDP * FDP, mlp_b1 + (size_t)l * FD,
                    nullptr, actB, N, Npad, 1, nullptr, nullptr, stream);
        launch_gemm(actB, wpM2 + (size_t)l * FDP * FDP, mlp_b2 + (size_t)l * FD,
                    buf, nullptr, N, Npad, 0, nullptr, nullptr, stream);

        // BatchNorm (batch stats) + relu -> h
        zero_kernel<<<3, BLK, 0, stream>>>(stats, 2 * FD);
        bn_reduce_kernel<<<512, 320, 0, stream>>>(buf, stats, N);
        bn_finalize_kernel<<<1, 320, 0, stream>>>(
            stats, bn_gamma + (size_t)l * FD, bn_beta + (size_t)l * FD, ss, N);
        bn_apply_kernel<<<(int)((ndTot + BLK - 1) / BLK), BLK, 0, stream>>>(
            buf, ss, h, ndTot);
    }

    // ---- mean pooling + final projection ----
    zero_kernel<<<(int)(((long)G * FD + G + BLK - 1) / BLK), BLK, 0, stream>>>(
        sums, (long)G * FD + G);
    pool_scatter_kernel<<<N, 320, 0, stream>>>(h, batch, sums, cnts, N);
    {
        long tot = (long)G * FDP;
        pool_mean_kernel<<<(int)((tot + BLK - 1) / BLK), BLK, 0, stream>>>(
            sums, cnts, actA, G);
    }
    launch_gemm(actA, wpPool, pool_b, out, nullptr, G, Gpad, 0,
                nullptr, nullptr, stream);
}
